// Transformer_8761733284341
// MI455X (gfx1250) — matrix operand order repaired and checked by reading, not compiled, not run on hardware
//
#include <hip/hip_runtime.h>
#include <hip/hip_fp16.h>

// ---------------- problem constants (match reference) ----------------
#define D_    50
#define S_    134
#define H_    5
#define DK_   10
#define L_    4
#define HID_  2048
#define B_    1024
#define R_    (B_ * S_)          // 137216 = 16 * 8576
#define NTILES_ (R_ / 16)        // 8576 row tiles
#define LDX_  64                 // padded activation leading dim (K padded to 64)
#define EPS_  1e-6f

typedef _Float16 v16h __attribute__((ext_vector_type(16)));
typedef float    v8f  __attribute__((ext_vector_type(8)));
typedef float    v4f  __attribute__((ext_vector_type(4)));

// ---------------- WMMA helper (gfx1250 wave32) ----------------
__device__ __forceinline__ v8f wmma16(v16h a, v16h b, v8f c) {
  // D = A(16x32 f16) * B(32x16 f16) + C(16x16 f32)
  return __builtin_amdgcn_wmma_f32_16x16x32_f16(
      /*neg_a=*/false, a, /*neg_b=*/false, b,
      /*c_mod=*/(short)0, c, /*reuse_a=*/false, /*reuse_b=*/false);
}

// A fragment from padded f32 activations [rows][LDX_], K in [k0, k0+32), no masks.
// Layout (ISA 7.12.2, 16-bit A): e -> K = k0 + 16*(e>>3) + 8*half + (e&7).
// Per lane this is two contiguous 8-float chunks (32B aligned) -> b128 loads.
__device__ __forceinline__ v16h load_a64(const float* __restrict__ src,
                                         int row0, int k0) {
  int lane = threadIdx.x & 31;
  int hf = lane >> 4, r = lane & 15;
  const float* p = src + ((size_t)(row0 + r) << 6) + k0 + 8 * hf;
  v4f c0 = *(const v4f*)(p);
  v4f c1 = *(const v4f*)(p + 4);
  v4f c2 = *(const v4f*)(p + 16);
  v4f c3 = *(const v4f*)(p + 20);
  v16h a;
#pragma unroll
  for (int j = 0; j < 4; ++j) {
    a[j]      = (_Float16)c0[j];
    a[4 + j]  = (_Float16)c1[j];
    a[8 + j]  = (_Float16)c2[j];
    a[12 + j] = (_Float16)c3[j];
  }
  return a;
}

// B fragment from transposed, zero-padded f16 weights W^T[Npad][Kpad].
// Layout: e -> K = k0 + 16*(e>>3) + 8*half + (e&7), N = col0 + (lane&15).
// Per lane: two contiguous 8-half runs (16B aligned).
__device__ __forceinline__ v16h load_bT(const _Float16* __restrict__ wt,
                                        int col0, int k0, int ldk) {
  int lane = threadIdx.x & 31;
  int hf = lane >> 4, r = lane & 15;
  const _Float16* p = wt + (size_t)(col0 + r) * ldk + k0 + 8 * hf;
  v16h b;
#pragma unroll
  for (int e = 0; e < 8; ++e) {
    b[e]     = p[e];
    b[8 + e] = p[16 + e];
  }
  return b;
}

// ---------------- weight transpose + pad + f32->f16 ----------------
// dst[n*Kpad + k] = (k<K && n<N) ? src[k*N + n] : 0
__global__ __launch_bounds__(256) void cvtT_kernel(
    const float* __restrict__ src, _Float16* __restrict__ dst,
    int K, int N, int Kpad, int total) {
  int idx = blockIdx.x * 256 + threadIdx.x;
  if (idx >= total) return;
  int n = idx / Kpad;
  int k = idx - n * Kpad;
  float v = (k < K && n < N) ? src[(size_t)k * N + n] : 0.f;
  dst[idx] = (_Float16)v;
}

// ---------------- embedding + positional encoding (padded layout) ----------------
__global__ __launch_bounds__(256) void embed_pe_kernel(
    const int* __restrict__ tokens, const float* __restrict__ emb,
    float* __restrict__ x) {
  int idx = blockIdx.x * 256 + threadIdx.x;
  if (idx >= R_ * LDX_) return;
  int d  = idx & (LDX_ - 1);
  int rs = idx >> 6;
  if (d >= D_) { x[idx] = 0.f; return; }
  int s   = rs % S_;
  int tok = tokens[rs];
  float div = expf(-(float)(d & ~1) * (9.210340371976184f / (float)D_));
  float ang = (float)s * div;
  float pe  = (d & 1) ? cosf(ang) : sinf(ang);
  x[idx] = emb[(size_t)tok * D_ + d] + pe;
}

// zero pad columns [50,64) of ctx once per call (attention writes only cols<50)
__global__ __launch_bounds__(256) void ctx_pad_zero_kernel(float* __restrict__ ctx) {
  int idx = blockIdx.x * 256 + threadIdx.x;
  if (idx >= R_ * (LDX_ - D_)) return;
  int row = idx / (LDX_ - D_);
  int c   = idx - row * (LDX_ - D_);
  ctx[(size_t)row * LDX_ + D_ + c] = 0.f;
}

// ---------------- QKV projection ----------------
__global__ __launch_bounds__(32) void qkv_kernel(
    const float* __restrict__ x,
    const _Float16* __restrict__ wqT, const _Float16* __restrict__ wkT,
    const _Float16* __restrict__ wvT,
    const float* __restrict__ bq, const float* __restrict__ bk,
    const float* __restrict__ bv,
    float* __restrict__ q, float* __restrict__ k, float* __restrict__ v) {
  int row0 = blockIdx.x * 16;
  int lane = threadIdx.x & 31;
  int hf = lane >> 4, r = lane & 15;

  v16h a0 = load_a64(x, row0, 0);
  v16h a1 = load_a64(x, row0, 32);

  const v8f fz = {0.f,0.f,0.f,0.f,0.f,0.f,0.f,0.f};
  v8f aq[4], ak[4], av[4];
#pragma unroll
  for (int nt = 0; nt < 4; ++nt) {
    int col0 = nt * 16;
    aq[nt] = wmma16(a1, load_bT(wqT, col0, 32, LDX_),
             wmma16(a0, load_bT(wqT, col0, 0,  LDX_), fz));
    ak[nt] = wmma16(a1, load_bT(wkT, col0, 32, LDX_),
             wmma16(a0, load_bT(wkT, col0, 0,  LDX_), fz));
    av[nt] = wmma16(a1, load_bT(wvT, col0, 32, LDX_),
             wmma16(a0, load_bT(wvT, col0, 0,  LDX_), fz));
  }
#pragma unroll
  for (int nt = 0; nt < 4; ++nt) {
    int col = nt * 16 + r;
    if (col < D_) {
      float bqv = bq[col], bkv = bk[col], bvv = bv[col];
#pragma unroll
      for (int i = 0; i < 8; ++i) {
        size_t off = (size_t)(row0 + i + 8*hf) * D_ + col;
        q[off] = aq[nt][i] + bqv;
        k[off] = ak[nt][i] + bkv;
        v[off] = av[nt][i] + bvv;
      }
    }
  }
}

// ---------------- attention: per (b,h) online-softmax (DK=10, VALU) ----------------
__global__ __launch_bounds__(128) void attn_kernel(
    const float* __restrict__ q, const float* __restrict__ k,
    const float* __restrict__ v, float* __restrict__ ctx) {
  __shared__ float qs[S_ * DK_];
  __shared__ float ksh[S_ * DK_];
  __shared__ float vsh[S_ * DK_];
  int bh = blockIdx.x;
  int b = bh / H_;
  int h = bh - b * H_;
  size_t base = (size_t)b * S_ * D_ + (size_t)h * DK_;

  for (int i = threadIdx.x; i < S_ * DK_; i += 128) {
    int s = i / DK_, d = i - s * DK_;
    size_t g = base + (size_t)s * D_ + d;
    qs[i]  = q[g];
    ksh[i] = k[g];
    vsh[i] = v[g];
  }
  __syncthreads();

  size_t obase = ((size_t)b * S_) * LDX_ + (size_t)h * DK_;
  const float scale = 0.31622776601683794f;  // 1/sqrt(10)
  for (int s = threadIdx.x; s < S_; s += 128) {
    float qr[DK_];
#pragma unroll
    for (int d = 0; d < DK_; ++d) qr[d] = qs[s * DK_ + d];
    float m = -3.0e38f, l = 0.f;
    float acc[DK_];
#pragma unroll
    for (int d = 0; d < DK_; ++d) acc[d] = 0.f;
    for (int j = 0; j < S_; ++j) {
      float t = 0.f;
#pragma unroll
      for (int d = 0; d < DK_; ++d) t += qr[d] * ksh[j * DK_ + d];
      t *= scale;
      float mn   = fmaxf(m, t);
      float corr = __expf(m - mn);
      float p    = __expf(t - mn);
      l = l * corr + p;
#pragma unroll
      for (int d = 0; d < DK_; ++d) acc[d] = acc[d] * corr + p * vsh[j * DK_ + d];
      m = mn;
    }
    float inv = 1.f / l;
#pragma unroll
    for (int d = 0; d < DK_; ++d)
      ctx[obase + (size_t)s * LDX_ + d] = acc[d] * inv;
  }
}

// ---------- output projection + residual + LayerNorm (in-place x) ----------
__global__ __launch_bounds__(32) void oproj_ln_kernel(
    const float* __restrict__ ctx, const _Float16* __restrict__ woT,
    const float* __restrict__ bo, const float* __restrict__ g,
    const float* __restrict__ bt, float* __restrict__ x) {
  __shared__ float tile[16][D_ + 2];
  int row0 = blockIdx.x * 16;
  int lane = threadIdx.x & 31;
  int hf = lane >> 4, r = lane & 15;

  v16h a0 = load_a64(ctx, row0, 0);
  v16h a1 = load_a64(ctx, row0, 32);
  const v8f fz = {0.f,0.f,0.f,0.f,0.f,0.f,0.f,0.f};
#pragma unroll
  for (int nt = 0; nt < 4; ++nt) {
    int col0 = nt * 16;
    v8f acc = wmma16(a1, load_bT(woT, col0, 32, LDX_),
              wmma16(a0, load_bT(woT, col0, 0,  LDX_), fz));
    int col = col0 + r;
    if (col < D_) {
      float bb = bo[col];
#pragma unroll
      for (int i = 0; i < 8; ++i) {
        int rl = i + 8*hf;
        tile[rl][col] = acc[i] + bb + x[((size_t)(row0 + rl) << 6) + col];
      }
    }
  }
  __syncthreads();
  if (lane < 16) {
    float m = 0.f;
#pragma unroll
    for (int c = 0; c < D_; ++c) m += tile[lane][c];
    m *= (1.f / D_);
    float var = 0.f;
#pragma unroll
    for (int c = 0; c < D_; ++c) { float dd = tile[lane][c] - m; var += dd * dd; }
    var *= (1.f / D_);
    float inv = rsqrtf(var + EPS_);
    size_t ro = (size_t)(row0 + lane) << 6;
#pragma unroll
    for (int c = 0; c < LDX_; ++c)
      x[ro + c] = (c < D_) ? ((tile[lane][c] - m) * inv * g[c] + bt[c]) : 0.f;
  }
}

// ---------- fused FFN (D->2048 relu ->D) + residual + LayerNorm (in-place) ----------
__global__ __launch_bounds__(32) void ffn_ln_kernel(
    const float* __restrict__ x, const _Float16* __restrict__ w1T,
    const float* __restrict__ b1, const _Float16* __restrict__ w2T,
    const float* __restrict__ b2, const float* __restrict__ g,
    const float* __restrict__ bt, float* __restrict__ xo) {
  // hidden chunk staged so the reader's A fragment is one contiguous 32B run
  __shared__ __align__(32) _Float16 hbuf[16 * 32];
  __shared__ float tile[16][D_ + 2];
  int row0 = blockIdx.x * 16;
  int lane = threadIdx.x & 31;
  int hf = lane >> 4, r = lane & 15;

  v16h a0 = load_a64(x, row0, 0);
  v16h a1 = load_a64(x, row0, 32);

  const v8f fz = {0.f,0.f,0.f,0.f,0.f,0.f,0.f,0.f};
  v8f acc[4];
#pragma unroll
  for (int nt = 0; nt < 4; ++nt) acc[nt] = fz;

  for (int hp = 0; hp < HID_; hp += 32) {
    if (hp + 32 < HID_) {  // gfx1250 global_prefetch_b8 for next weight chunk
      __builtin_prefetch(&w1T[(size_t)(hp + 32) * LDX_], 0, 1);
    }
    // h[16 x 32] = relu(x @ W1[:, hp:hp+32] + b1)
#pragma unroll
    for (int sub = 0; sub < 2; ++sub) {
      int colg0 = hp + sub * 16;
      v8f hc = wmma16(a1, load_bT(w1T, colg0, 32, LDX_),
               wmma16(a0, load_bT(w1T, colg0, 0,  LDX_), fz));
      float bb = b1[colg0 + r];
      // permuted store so reader lane (hf,r) sees elements e=0..15 contiguous:
      // addr(row,col) = row*32 + 16*((col>>3)&1) + 8*(col>>4) + (col&7)
      int col = sub * 16 + r;
      int wbase = 16 * ((col >> 3) & 1) + 8 * (col >> 4) + (col & 7);
#pragma unroll
      for (int i = 0; i < 8; ++i) {
        float t = hc[i] + bb;
        t = fmaxf(t, 0.f);
        hbuf[(i + 8*hf) * 32 + wbase] = (_Float16)t;
      }
    }
    __syncthreads();
    v16h ah = *(const v16h*)(hbuf + r * 32 + 16 * hf);   // 2x ds_load_b128
#pragma unroll
    for (int nt = 0; nt < 4; ++nt)
      acc[nt] = wmma16(ah, load_bT(w2T, nt * 16, hp, HID_), acc[nt]);
    __syncthreads();
  }

  // epilogue: + b2 + residual, LayerNorm, write back (pads = 0)
#pragma unroll
  for (int nt = 0; nt < 4; ++nt) {
    int col = nt * 16 + r;
    if (col < D_) {
      float bb = b2[col];
#pragma unroll
      for (int i = 0; i < 8; ++i) {
        int rl = i + 8*hf;
        tile[rl][col] = acc[nt][i] + bb + x[((size_t)(row0 + rl) << 6) + col];
      }
    }
  }
  __syncthreads();
  if (lane < 16) {
    float m = 0.f;
#pragma unroll
    for (int c = 0; c < D_; ++c) m += tile[lane][c];
    m *= (1.f / D_);
    float var = 0.f;
#pragma unroll
    for (int c = 0; c < D_; ++c) { float dd = tile[lane][c] - m; var += dd * dd; }
    var *= (1.f / D_);
    float inv = rsqrtf(var + EPS_);
    size_t ro = (size_t)(row0 + lane) << 6;
#pragma unroll
    for (int c = 0; c < LDX_; ++c)
      xo[ro + c] = (c < D_) ? ((tile[lane][c] - m) * inv * g[c] + bt[c]) : 0.f;
  }
}

// ---------- final LN + lin1(D->1) + lin2(S->1) + sigmoid ----------
__global__ __launch_bounds__(128) void head_kernel(
    const float* __restrict__ x, const float* __restrict__ lnf_g,
    const float* __restrict__ lnf_b, const float* __restrict__ w1,
    const float* __restrict__ w1b, const float* __restrict__ w2,
    const float* __restrict__ w2b, float* __restrict__ out) {
  __shared__ float red[128];
  int b = blockIdx.x;
  float part = 0.f;
  for (int s = threadIdx.x; s < S_; s += 128) {
    const float* row = x + (((size_t)b * S_ + s) << 6);
    float m = 0.f;
#pragma unroll
    for (int c = 0; c < D_; ++c) m += row[c];
    m *= (1.f / D_);
    float var = 0.f;
#pragma unroll
    for (int c = 0; c < D_; ++c) { float dd = row[c] - m; var += dd * dd; }
    var *= (1.f / D_);
    float inv = rsqrtf(var + EPS_);
    float dot = 0.f;
#pragma unroll
    for (int c = 0; c < D_; ++c)
      dot += ((row[c] - m) * inv * lnf_g[c] + lnf_b[c]) * w1[c];
    part += (dot + w1b[0]) * w2[s];
  }
  red[threadIdx.x] = part;
  __syncthreads();
  for (int off = 64; off > 0; off >>= 1) {
    if (threadIdx.x < off) red[threadIdx.x] += red[threadIdx.x + off];
    __syncthreads();
  }
  if (threadIdx.x == 0)
    out[b] = 1.f / (1.f + __expf(-(red[0] + w2b[0])));
}

// ---------------- host launcher ----------------
extern "C" void kernel_launch(void* const* d_in, const int* in_sizes, int n_in,
                              void* d_out, int out_size, void* d_ws, size_t ws_size,
                              hipStream_t stream) {
  (void)in_sizes; (void)n_in; (void)out_size; (void)ws_size;
  const int*   tokens = (const int*)  d_in[0];
  const float* emb    = (const float*)d_in[1];
  const float* Wq = (const float*)d_in[2];   const float* bq = (const float*)d_in[3];
  const float* Wk = (const float*)d_in[4];   const float* bk = (const float*)d_in[5];
  const float* Wv = (const float*)d_in[6];   const float* bv = (const float*)d_in[7];
  const float* Wo = (const float*)d_in[8];   const float* bo = (const float*)d_in[9];
  const float* ln1_g = (const float*)d_in[10]; const float* ln1_b = (const float*)d_in[11];
  const float* W1 = (const float*)d_in[12];  const float* b1 = (const float*)d_in[13];
  const float* W2 = (const float*)d_in[14];  const float* b2 = (const float*)d_in[15];
  const float* ln2_g = (const float*)d_in[16]; const float* ln2_b = (const float*)d_in[17];
  const float* lnf_g = (const float*)d_in[18]; const float* lnf_b = (const float*)d_in[19];
  const float* l1w = (const float*)d_in[20]; const float* l1b = (const float*)d_in[21];
  const float* l2w = (const float*)d_in[22]; const float* l2b = (const float*)d_in[23];
  float* out = (float*)d_out;

  // workspace layout
  const size_t RX = (size_t)R_ * LDX_;  // padded activations
  const size_t RQ = (size_t)R_ * D_;    // q/k/v (VALU consumers)
  float* x   = (float*)d_ws;
  float* ctx = x + RX;
  float* q   = ctx + RX;
  float* k   = q + RQ;
  float* v   = k + RQ;
  _Float16* WqT = (_Float16*)(v + RQ);            // [L][64][64]
  _Float16* WkT = WqT + (size_t)L_ * LDX_ * LDX_;
  _Float16* WvT = WkT + (size_t)L_ * LDX_ * LDX_;
  _Float16* WoT = WvT + (size_t)L_ * LDX_ * LDX_;
  _Float16* W1T = WoT + (size_t)L_ * LDX_ * LDX_; // [L][2048][64]
  _Float16* W2T = W1T + (size_t)L_ * HID_ * LDX_; // [L][64][2048]

  auto cvtT = [&](const float* s, _Float16* d, int K, int N, int Kpad, int Npad) {
    int total = Npad * Kpad;
    cvtT_kernel<<<(total + 255) / 256, 256, 0, stream>>>(s, d, K, N, Kpad, total);
  };
  for (int l = 0; l < L_; ++l) {
    cvtT(Wq + (size_t)l * D_ * D_,  WqT + (size_t)l * LDX_ * LDX_, D_,   D_,   LDX_, LDX_);
    cvtT(Wk + (size_t)l * D_ * D_,  WkT + (size_t)l * LDX_ * LDX_, D_,   D_,   LDX_, LDX_);
    cvtT(Wv + (size_t)l * D_ * D_,  WvT + (size_t)l * LDX_ * LDX_, D_,   D_,   LDX_, LDX_);
    cvtT(Wo + (size_t)l * D_ * D_,  WoT + (size_t)l * LDX_ * LDX_, D_,   D_,   LDX_, LDX_);
    cvtT(W1 + (size_t)l * D_ * HID_, W1T + (size_t)l * HID_ * LDX_, D_,  HID_, LDX_, HID_);
    cvtT(W2 + (size_t)l * HID_ * D_, W2T + (size_t)l * LDX_ * HID_, HID_, D_,  HID_, LDX_);
  }

  embed_pe_kernel<<<(R_ * LDX_ + 255) / 256, 256, 0, stream>>>(tokens, emb, x);
  ctx_pad_zero_kernel<<<(R_ * (LDX_ - D_) + 255) / 256, 256, 0, stream>>>(ctx);

  for (int l = 0; l < L_; ++l) {
    qkv_kernel<<<NTILES_, 32, 0, stream>>>(
        x, WqT + (size_t)l * LDX_ * LDX_, WkT + (size_t)l * LDX_ * LDX_,
        WvT + (size_t)l * LDX_ * LDX_, bq + l * D_, bk + l * D_, bv + l * D_,
        q, k, v);
    attn_kernel<<<B_ * H_, 128, 0, stream>>>(q, k, v, ctx);
    oproj_ln_kernel<<<NTILES_, 32, 0, stream>>>(
        ctx, WoT + (size_t)l * LDX_ * LDX_, bo + l * D_,
        ln1_g + l * D_, ln1_b + l * D_, x);
    ffn_ln_kernel<<<NTILES_, 32, 0, stream>>>(
        x, W1T + (size_t)l * HID_ * LDX_, b1 + l * HID_,
        W2T + (size_t)l * LDX_ * HID_, b2 + l * D_,
        ln2_g + l * D_, ln2_b + l * D_, x);
  }

  head_kernel<<<B_, 128, 0, stream>>>(x, lnf_g, lnf_b, l1w, l1b, l2w, l2b, out);
}
